// CONTime_59596966199457
// MI455X (gfx1250) — matrix operand order repaired and checked by reading, not compiled, not run on hardware
//
#include <hip/hip_runtime.h>
#include <hip/hip_bf16.h>

// ---------------------------------------------------------------------------
// Neural-CDE scan for MI455X (gfx1250, wave32, WMMA).
//
//   build_xk   : spline knots Xk (B,L,C) from coeffs            [parallel]
//   init_h0    : h0 = Xk[:,0]@W_init+b ; rev_h0 = Xk[:,-1]@W_rev [parallel]
//   pack_w1/w2 : f32->f16 repack into WMMA B-operand lane layout [parallel]
//   scan       : 16 blocks = 8 batch-groups x 2 directions; each block runs
//                its sequential scan with v_wmma_f32_16x16x32_f16, LDS state,
//                workgroup barriers. Forward 1023 steps, reverse only 335
//                (only rh[0:336] contributes to the output window).
//   pred/dyhat : tiny (.,128)@(128,16) readouts.
// ---------------------------------------------------------------------------

typedef __attribute__((ext_vector_type(16))) _Float16 v16h;
typedef __attribute__((ext_vector_type(8)))  float    v8f;

#define B_TOT   128
#define L_SEQ   1024
#define C_DIM   16
#define H_DIM   128
#define FH_DIM  128
#define PRED    336
#define NSTEP_F (L_SEQ - 1)      // 1023 forward steps
#define NSTEP_R (PRED - 1)       // 335 reverse steps needed

// workspace layout (bytes, all 32B aligned)
#define XK_BYTES   ((size_t)B_TOT * L_SEQ * C_DIM * 4)      // 8 MiB
#define XK_OFF     ((size_t)0)
#define H0_OFF     (XK_OFF + XK_BYTES)
#define H0_BYTES   ((size_t)B_TOT * H_DIM * 4)
#define RH0_OFF    (H0_OFF + H0_BYTES)
#define W1P_OFF    (RH0_OFF + H0_BYTES)
#define W1P_ELTS   (8 * 5 * 32 * 16)                         // ntile x kchunk x lane x 16
#define W2P_OFF    (W1P_OFF + (size_t)W1P_ELTS * 2)
#define W2P_ELTS   (128 * 4 * 32 * 16)
#define HF_OFF     (W2P_OFF + (size_t)W2P_ELTS * 2)
#define HBUF_BYTES ((size_t)PRED * B_TOT * H_DIM * 4)        // 21 MiB each
#define HR_OFF     (HF_OFF + HBUF_BYTES)
// total ws usage ~53 MiB

__device__ __forceinline__ float fast_tanh(float x) {
#if __has_builtin(__builtin_amdgcn_tanhf)
  return __builtin_amdgcn_tanhf(x);
#else
  return tanhf(x);
#endif
}

__device__ __forceinline__ v8f wmma_f16(v16h a, v16h b, v8f c) {
  // (neg_a, A, neg_b, B, c_mod, C, reuse_a, reuse_b)
  return __builtin_amdgcn_wmma_f32_16x16x32_f16(false, a, false, b, (short)0, c,
                                                false, false);
}

// --- Xk = spline knots --------------------------------------------------
__global__ void build_xk_kernel(const float* __restrict__ coeffs,
                                float* __restrict__ Xk) {
  int i = blockIdx.x * blockDim.x + threadIdx.x;
  if (i >= B_TOT * L_SEQ * C_DIM) return;
  int c = i & 15;
  int t = (i >> 4) & (L_SEQ - 1);
  int b = i >> 14;
  float v;
  if (t < L_SEQ - 1) {
    v = coeffs[((size_t)b * (L_SEQ - 1) + t) * 64 + c];               // a-part
  } else {
    size_t base = ((size_t)b * (L_SEQ - 1) + (L_SEQ - 2)) * 64;
    v = coeffs[base + c] + coeffs[base + 16 + c] +
        coeffs[base + 32 + c] + coeffs[base + 48 + c];                // a+b+c+d
  }
  Xk[i] = v;
}

// --- h0 / rev_h0 --------------------------------------------------------
__global__ void init_h0_kernel(const float* __restrict__ Xk,
                               const float* __restrict__ Wi, const float* __restrict__ bi,
                               const float* __restrict__ Wr, const float* __restrict__ br,
                               float* __restrict__ h0, float* __restrict__ rh0) {
  int i = blockIdx.x * blockDim.x + threadIdx.x;
  if (i >= B_TOT * H_DIM) return;
  int j = i & (H_DIM - 1);
  int b = i >> 7;
  const float* x0 = &Xk[((size_t)b * L_SEQ + 0) * C_DIM];
  const float* xl = &Xk[((size_t)b * L_SEQ + (L_SEQ - 1)) * C_DIM];
  float s0 = bi[j], s1 = br[j];
#pragma unroll
  for (int c = 0; c < C_DIM; ++c) {
    s0 += x0[c] * Wi[c * H_DIM + j];
    s1 += xl[c] * Wr[c * H_DIM + j];
  }
  h0[i] = s0;
  rh0[i] = s1;
}

// --- Pack W1 (144x128, K padded to 160) into WMMA B-operand layout ------
// B layout (16x16x32 f16): lane group p=lane/16 holds K = 32*kc + j + 8*(j>>3) + 8*p.
__global__ void pack_w1_kernel(const float* __restrict__ W1, _Float16* __restrict__ out) {
  int i = blockIdx.x * blockDim.x + threadIdx.x;
  if (i >= W1P_ELTS) return;
  int j  = i & 15;
  int ln = (i >> 4) & 31;
  int r2 = i >> 9;
  int kc = r2 % 5;
  int nt = r2 / 5;
  int K  = 32 * kc + j + 8 * (j >> 3) + 8 * (ln >> 4);
  int n  = nt * 16 + (ln & 15);
  out[i] = (K < (C_DIM + H_DIM)) ? (_Float16)W1[K * FH_DIM + n] : (_Float16)0.0f;
}

// --- Pack W2 (128x2048) ------------------------------------------------
__global__ void pack_w2_kernel(const float* __restrict__ W2, _Float16* __restrict__ out) {
  int i = blockIdx.x * blockDim.x + threadIdx.x;
  if (i >= W2P_ELTS) return;
  int j  = i & 15;
  int ln = (i >> 4) & 31;
  int kc = (i >> 9) & 3;
  int nt = i >> 11;
  int K  = 32 * kc + j + 8 * (j >> 3) + 8 * (ln >> 4);
  int n  = nt * 16 + (ln & 15);
  out[i] = (_Float16)W2[(size_t)K * 2048 + n];
}

// --- The sequential CDE scan -------------------------------------------
// grid = 16 blocks: blockIdx>>3 = direction, blockIdx&7 = batch group of 16.
// block = 512 threads (16 waves).  LDS ~60 KB.
__global__ __launch_bounds__(512, 1)
void scan_kernel(const float* __restrict__ Xk,
                 const float* __restrict__ h0g, const float* __restrict__ rh0g,
                 const _Float16* __restrict__ w1p, const _Float16* __restrict__ w2p,
                 const float* __restrict__ b1, const float* __restrict__ b2,
                 float* __restrict__ hbuf_f, float* __restrict__ hbuf_r) {
  __shared__ __align__(32) _Float16 sW1[W1P_ELTS];   // 40960 B, resident all scan
  __shared__ __align__(32) _Float16 sZ[5 * 32 * 16];   // A-packed z (16 x 160)
  __shared__ __align__(32) _Float16 sH1[4 * 32 * 16];  // A-packed h1 (16 x 128)
  __shared__ float sH[16 * H_DIM];                     // f32 hidden state
  __shared__ float sXn[16 * C_DIM];
  __shared__ float sDX[16 * C_DIM];
  __shared__ float sB1[FH_DIM];

  const int tid  = threadIdx.x;
  const int wave = tid >> 5;
  const int lane = tid & 31;
  const int lo   = lane & 15;
  const int hi   = lane >> 4;
  const int dir  = blockIdx.x >> 3;
  const int b0   = (blockIdx.x & 7) * 16;
  const int nstep = dir ? NSTEP_R : NSTEP_F;
  const float* h0 = dir ? rh0g : h0g;
  float* hbuf = dir ? hbuf_r : hbuf_f;

  for (int i = tid; i < W1P_ELTS; i += 512) sW1[i] = w1p[i];
  for (int i = tid; i < FH_DIM; i += 512) sB1[i] = b1[i];
  for (int i = tid; i < 16 * H_DIM; i += 512)
    sH[i] = h0[(size_t)(b0 + (i >> 7)) * H_DIM + (i & 127)];
  __syncthreads();

  // z0 = [zeros(C), h0], A-packed: element j of lane holds
  // K = 32*kc + j + 8*(j>>3) + 8*(lane>>4), row = lane&15
  for (int i = tid; i < 5 * 32 * 16; i += 512) {
    int j = i & 15, ln = (i >> 4) & 31, kc = i >> 9;
    int K = 32 * kc + j + 8 * (j >> 3) + 8 * (ln >> 4);
    int row = ln & 15;
    float val = (K >= C_DIM && K < C_DIM + H_DIM) ? sH[row * H_DIM + (K - C_DIM)] : 0.0f;
    sZ[i] = (_Float16)val;
  }
  if (dir) {  // rh index 0 = rev_h0 contributes to the output window
    float* dst = hbuf + (size_t)b0 * H_DIM;
    for (int i = tid; i < 16 * H_DIM; i += 512) dst[i] = sH[i];
  }
  __syncthreads();

  // ---- loop-invariant per-thread bases (keeps the hot loop spill-free) ----
  // P1: thread tid<256 owns one (row r = tid>>4, channel c = tid&15).
  const float* xrow =
      Xk + (size_t)(b0 + (tid >> 4)) * L_SEQ * C_DIM + (tid & 15);
  // P2: h1 scatter base; n = wave*16+lo is fixed per lane.
  const int n_p2 = wave * 16 + lo;
  const int kc2 = n_p2 >> 5, k2 = n_p2 & 31;
  const int p2 = (k2 >> 3) & 1;
  const int j2 = (k2 & 7) + 8 * (k2 >> 4);
  _Float16* h1dst = &sH1[(kc2 * 32 + 16 * p2 + 8 * hi) * 16 + j2]; // +v*16 per row
  // P3: W2 / b2 bases; this wave owns N-tiles nt = wave*8 .. wave*8+7.
  const _Float16* w2base = w2p + (size_t)(wave * 8) * 2048 + lane * 16;
  const float* b2base = b2 + wave * 8 * 16 + lo;
  float* dstBase = hbuf + (size_t)b0 * H_DIM;

  for (int t = 0; t < nstep; ++t) {
    // P1: dX, Xn for this step (256 threads, one element each)
    int tiN = dir ? (L_SEQ - 2 - t) : (t + 1);
    int tiP = dir ? (L_SEQ - 1 - t) : t;
    if (tid < 256) {
      float xn = xrow[(size_t)tiN * C_DIM];
      float xp = xrow[(size_t)tiP * C_DIM];
      sXn[tid] = xn;
      sDX[tid] = xn - xp;
    }
    // P2: h1 = tanh(z @ W1 + b1). 8 N-tiles -> waves 0..7, K=160 (5 chunks)
    if (wave < 8) {
      v8f acc = {};
#pragma unroll
      for (int kc = 0; kc < 5; ++kc) {
        v16h a = *(const v16h*)&sZ[(kc * 32 + lane) * 16];
        v16h b = *(const v16h*)&sW1[((wave * 5 + kc) * 32 + lane) * 16];
        acc = wmma_f16(a, b, acc);
      }
      float bv = sB1[n_p2];
#pragma unroll
      for (int v = 0; v < 8; ++v) {
        float hv = fast_tanh(acc[v] + bv);
        h1dst[v * 16] = (_Float16)hv;   // A-packed scatter, stride 32B
      }
    }
    __syncthreads();
    // P3: mat = tanh(h1 @ W2 + b2); h += sum_c mat[:,j,c]*dX[:,c]
    float dxv[8];
#pragma unroll
    for (int v = 0; v < 8; ++v) dxv[v] = sDX[((v + 8 * hi) << 4) | lo];
    v16h a0 = *(const v16h*)&sH1[(0 * 32 + lane) * 16];
    v16h a1 = *(const v16h*)&sH1[(1 * 32 + lane) * 16];
    v16h a2 = *(const v16h*)&sH1[(2 * 32 + lane) * 16];
    v16h a3 = *(const v16h*)&sH1[(3 * 32 + lane) * 16];
#pragma unroll 2
    for (int i = 0; i < 8; ++i) {
      const _Float16* wb = w2base + (size_t)i * 2048;
      v8f acc = {};
      acc = wmma_f16(a0, *(const v16h*)&wb[0 * 512], acc);
      acc = wmma_f16(a1, *(const v16h*)&wb[1 * 512], acc);
      acc = wmma_f16(a2, *(const v16h*)&wb[2 * 512], acc);
      acc = wmma_f16(a3, *(const v16h*)&wb[3 * 512], acc);
      float b2v = b2base[i * 16];
      float red[8];
#pragma unroll
      for (int v = 0; v < 8; ++v) {
        // C/D layout: lane holds col c=lo, row M=v+8*hi
        float m = fast_tanh(acc[v] + b2v) * dxv[v];
        m += __shfl_xor(m, 1, 32);
        m += __shfl_xor(m, 2, 32);
        m += __shfl_xor(m, 4, 32);
        m += __shfl_xor(m, 8, 32);
        red[v] = m;   // after butterfly: identical across the 16-lane group
      }
      // select red[lo&7] with a static tree, then 16 lanes do 1 RMW each
      float r01 = (lo & 1) ? red[1] : red[0];
      float r23 = (lo & 1) ? red[3] : red[2];
      float r45 = (lo & 1) ? red[5] : red[4];
      float r67 = (lo & 1) ? red[7] : red[6];
      float r03 = (lo & 2) ? r23 : r01;
      float r47 = (lo & 2) ? r67 : r45;
      float rv  = (lo & 4) ? r47 : r03;
      int nt = wave * 8 + i;
      if (lo < 8) sH[(lo + 8 * hi) * H_DIM + nt] += rv;
    }
    __syncthreads();
    // P4: emit h slice for the output window
    bool doStore = dir ? true : (t >= NSTEP_F - PRED);
    int slot = dir ? (t + 1) : (t - (NSTEP_F - PRED));
    if (doStore) {
      float* dst = dstBase + (size_t)slot * (B_TOT * H_DIM);
      for (int i = tid; i < 16 * H_DIM; i += 512) dst[i] = sH[i];
    }
    // P5: z <- [Xn, h] (A-packed, K padded with zeros)
    for (int i = tid; i < 5 * 32 * 16; i += 512) {
      int j = i & 15, ln = (i >> 4) & 31, kc = i >> 9;
      int K = 32 * kc + j + 8 * (j >> 3) + 8 * (ln >> 4);
      int row = ln & 15;
      float val;
      if (K < C_DIM)               val = sXn[(row << 4) | K];
      else if (K < C_DIM + H_DIM)  val = sH[row * H_DIM + (K - C_DIM)];
      else                         val = 0.0f;
      sZ[i] = (_Float16)val;
    }
    __syncthreads();
  }
}

// --- pred_y = (h_fwd[s] + h_rev[335-s]) @ W_read + b_read ---------------
__global__ __launch_bounds__(256)
void pred_kernel(const float* __restrict__ hf, const float* __restrict__ hr,
                 const float* __restrict__ Wread, const float* __restrict__ bread,
                 float* __restrict__ out) {
  __shared__ float sh[16 * H_DIM];
  __shared__ float sw[H_DIM * C_DIM];
  int tid = threadIdx.x;
  for (int i = tid; i < H_DIM * C_DIM; i += 256) sw[i] = Wread[i];
  int row0 = blockIdx.x * 16;
  for (int i = tid; i < 16 * H_DIM; i += 256) {
    int rl = i >> 7, k = i & 127;
    int row = row0 + rl;
    int b = row / PRED, s = row % PRED;
    sh[i] = hf[((size_t)s * B_TOT + b) * H_DIM + k] +
            hr[((size_t)(PRED - 1 - s) * B_TOT + b) * H_DIM + k];
  }
  __syncthreads();
  int rl = tid >> 4, c = tid & 15;
  float s = bread[c];
#pragma unroll 8
  for (int k = 0; k < H_DIM; ++k) s += sh[rl * H_DIM + k] * sw[k * C_DIM + c];
  out[(size_t)(row0 + rl) * C_DIM + c] = s;
}

// --- dyhat_dt = (dhdt[689+s, b] @ W_read) ------------------------------
__global__ __launch_bounds__(256)
void dyhat_kernel(const float* __restrict__ dhdt, const float* __restrict__ Wread,
                  float* __restrict__ out) {
  __shared__ float sh[16 * H_DIM];
  __shared__ float sw[H_DIM * C_DIM];
  int tid = threadIdx.x;
  for (int i = tid; i < H_DIM * C_DIM; i += 256) sw[i] = Wread[i];
  int row0 = blockIdx.x * 16;
  const int NROW = B_TOT * (PRED - 1);
  for (int i = tid; i < 16 * H_DIM; i += 256) {
    int rl = i >> 7, k = i & 127;
    int row = row0 + rl;
    if (row < NROW) {
      int b = row / (PRED - 1), s = row % (PRED - 1);
      int tt = L_SEQ - (PRED - 1) + s;  // 689 + s
      sh[i] = dhdt[((size_t)tt * B_TOT + b) * H_DIM + k];
    } else {
      sh[i] = 0.0f;
    }
  }
  __syncthreads();
  int rl = tid >> 4, c = tid & 15;
  int row = row0 + rl;
  if (row < NROW) {
    float s = 0.0f;
#pragma unroll 8
    for (int k = 0; k < H_DIM; ++k) s += sh[rl * H_DIM + k] * sw[k * C_DIM + c];
    out[(size_t)row * C_DIM + c] = s;
  }
}

extern "C" void kernel_launch(void* const* d_in, const int* in_sizes, int n_in,
                              void* d_out, int out_size, void* d_ws, size_t ws_size,
                              hipStream_t stream) {
  (void)in_sizes; (void)n_in; (void)out_size; (void)ws_size;
  // setup_inputs order: x, coeffs, times, dhdt, W_init, b_init, W_rev, b_rev,
  //                     W_read, b_read, W1, b1, W2, b2   (x, times unused)
  const float* coeffs = (const float*)d_in[1];
  const float* dhdt   = (const float*)d_in[3];
  const float* W_init = (const float*)d_in[4];
  const float* b_init = (const float*)d_in[5];
  const float* W_rev  = (const float*)d_in[6];
  const float* b_rev  = (const float*)d_in[7];
  const float* W_read = (const float*)d_in[8];
  const float* b_read = (const float*)d_in[9];
  const float* W1     = (const float*)d_in[10];
  const float* b1     = (const float*)d_in[11];
  const float* W2     = (const float*)d_in[12];
  const float* b2     = (const float*)d_in[13];

  char* ws = (char*)d_ws;   // uses ~53 MiB of scratch
  float*    Xk  = (float*)(ws + XK_OFF);
  float*    h0  = (float*)(ws + H0_OFF);
  float*    rh0 = (float*)(ws + RH0_OFF);
  _Float16* w1p = (_Float16*)(ws + W1P_OFF);
  _Float16* w2p = (_Float16*)(ws + W2P_OFF);
  float*    hf  = (float*)(ws + HF_OFF);
  float*    hr  = (float*)(ws + HR_OFF);
  float*    out = (float*)d_out;

  build_xk_kernel<<<(B_TOT * L_SEQ * C_DIM + 255) / 256, 256, 0, stream>>>(coeffs, Xk);
  init_h0_kernel<<<(B_TOT * H_DIM + 255) / 256, 256, 0, stream>>>(Xk, W_init, b_init,
                                                                  W_rev, b_rev, h0, rh0);
  pack_w1_kernel<<<(W1P_ELTS + 255) / 256, 256, 0, stream>>>(W1, w1p);
  pack_w2_kernel<<<(W2P_ELTS + 255) / 256, 256, 0, stream>>>(W2, w2p);
  scan_kernel<<<16, 512, 0, stream>>>(Xk, h0, rh0, w1p, w2p, b1, b2, hf, hr);
  pred_kernel<<<(B_TOT * PRED) / 16, 256, 0, stream>>>(hf, hr, W_read, b_read, out);
  dyhat_kernel<<<(B_TOT * (PRED - 1) + 15) / 16, 256, 0, stream>>>(
      dhdt, W_read, out + (size_t)B_TOT * PRED * C_DIM);
}
